// MemaBlock_12592844112348
// MI455X (gfx1250) — matrix operand order repaired and checked by reading, not compiled, not run on hardware
//
#include <hip/hip_runtime.h>
#include <hip/hip_bf16.h>
#include <math.h>

typedef __bf16 bf16;
typedef __attribute__((ext_vector_type(16))) __bf16 v16bf;
typedef __attribute__((ext_vector_type(8)))  __bf16 v8bf;
typedef __attribute__((ext_vector_type(8)))  float  v8f;

#define BB    2
#define TT    2048
#define DDIM  1024
#define NHEAD 16
#define HDIM  64
#define NDIMC 16
#define HIDC  4096

#define WMMA_BF16(A_, B_, C_) \
  __builtin_amdgcn_wmma_f32_16x16x32_bf16(false, (A_), false, (B_), (short)0, (C_), false, false)

static __device__ __forceinline__ v16bf cat8(v8bf lo, v8bf hi) {
  return __builtin_shufflevector(lo, hi, 0, 1, 2, 3, 4, 5, 6, 7,
                                 8, 9, 10, 11, 12, 13, 14, 15);
}

static __device__ __forceinline__ float gelu_tanh(float x) {
  float x3 = x * x * x;
  return 0.5f * x * (1.0f + tanhf(0.7978845608028654f * (x + 0.044715f * x3)));
}

// Async global->LDS 16B copy per lane (CDNA5, tracked by ASYNCcnt).
// INST_OFFSET is added to BOTH the LDS and global address.
static __device__ __forceinline__ void async_copy16(unsigned lds_off, const void* gaddr) {
  asm volatile("global_load_async_to_lds_b128 %0, %1, off"
               :: "v"(lds_off), "v"(gaddr) : "memory");
}
static __device__ __forceinline__ void async_copy16_off16(unsigned lds_off, const void* gaddr) {
  asm volatile("global_load_async_to_lds_b128 %0, %1, off offset:16"
               :: "v"(lds_off), "v"(gaddr) : "memory");
}
static __device__ __forceinline__ void wait_async0() {
  asm volatile("s_wait_asynccnt 0x0" ::: "memory");
}
static __device__ __forceinline__ unsigned lds_addr(const void* p) {
  return (unsigned)(unsigned long long)p;   // low 32 bits = LDS offset
}

// ---------------------------------------------------------------------------
// Tiled transpose + convert: in (K x N f32, row-major) -> out (N x K bf16)
// ---------------------------------------------------------------------------
__global__ __launch_bounds__(256)
void transpose_f32_to_bf16(const float* __restrict__ in, bf16* __restrict__ out,
                           int K, int N) {
  __shared__ float tile[32][33];
  int kb = blockIdx.y * 32, nb = blockIdx.x * 32;
  int tx = threadIdx.x & 31, ty = threadIdx.x >> 5;   // 32 x 8
#pragma unroll
  for (int i = ty; i < 32; i += 8)
    tile[i][tx] = in[(size_t)(kb + i) * N + nb + tx];
  __syncthreads();
#pragma unroll
  for (int i = ty; i < 32; i += 8)
    out[(size_t)(nb + i) * K + kb + tx] = (bf16)tile[tx][i];
}

// ---------------------------------------------------------------------------
// RMSNorm over D=1024; one block per row; optional f32 and bf16 outputs
// ---------------------------------------------------------------------------
__global__ __launch_bounds__(256)
void rmsnorm_kernel(const float* __restrict__ x, const float* __restrict__ g,
                    float* __restrict__ outf, bf16* __restrict__ outb) {
  __shared__ float red[8];
  __shared__ float rinv_s;
  int row = blockIdx.x;
  int tid = threadIdx.x, lane = tid & 31, wv = tid >> 5;
  const float* xr = x + (size_t)row * DDIM;
  float ss = 0.0f;
  for (int i = tid; i < DDIM; i += 256) { float v = xr[i]; ss += v * v; }
#pragma unroll
  for (int off = 16; off > 0; off >>= 1) ss += __shfl_xor(ss, off, 32);
  if (lane == 0) red[wv] = ss;
  __syncthreads();
  if (tid == 0) {
    float s = 0.0f;
    for (int i = 0; i < 8; ++i) s += red[i];
    rinv_s = rsqrtf(s / (float)DDIM + 1e-6f);
  }
  __syncthreads();
  float rinv = rinv_s;
  for (int i = tid; i < DDIM; i += 256) {
    float v = xr[i] * rinv * g[i];
    if (outf) outf[(size_t)row * DDIM + i] = v;
    if (outb) outb[(size_t)row * DDIM + i] = (bf16)v;
  }
}

// ---------------------------------------------------------------------------
// EMA scan: one thread per (b, d); 16 states per thread, scan over T.
// ---------------------------------------------------------------------------
__global__ __launch_bounds__(256)
void ema_kernel(const float* __restrict__ xn, const float* __restrict__ alpha_p,
                const float* __restrict__ delta_p, const float* __restrict__ beta,
                const float* __restrict__ eta, bf16* __restrict__ y) {
  int idx = blockIdx.x * 256 + threadIdx.x;
  if (idx >= BB * DDIM) return;
  int b = idx / DDIM, d = idx % DDIM;
  float a[NDIMC], dec[NDIMC], bet[NDIMC], et[NDIMC], h[NDIMC];
#pragma unroll
  for (int i = 0; i < NDIMC; ++i) {
    float av = 1.0f / (1.0f + expf(-alpha_p[d * NDIMC + i]));
    float dv = 1.0f / (1.0f + expf(-delta_p[d * NDIMC + i]));
    a[i] = av; dec[i] = 1.0f - av * dv;
    bet[i] = beta[d * NDIMC + i]; et[i] = eta[d * NDIMC + i];
    h[i] = 0.0f;
  }
  for (int t = 0; t < TT; ++t) {
    float xv = xn[((size_t)b * TT + t) * DDIM + d];
    float yv = 0.0f;
#pragma unroll
    for (int i = 0; i < NDIMC; ++i) {
      h[i] = a[i] * (xv * bet[i]) + dec[i] * h[i];
      yv += h[i] * et[i];
    }
    y[((size_t)b * TT + t) * DDIM + d] = (bf16)yv;
  }
}

// ---------------------------------------------------------------------------
// RoPE on first 32 dims of each 64-wide head.
// vt==0: out (B,H,T,64).  vt==1: out (B,H,64,T)  (channel-major, for AV tiles)
// ---------------------------------------------------------------------------
__global__ __launch_bounds__(256)
void rope_kernel(const bf16* __restrict__ in, bf16* __restrict__ out, int vt) {
  int idx = blockIdx.x * 256 + threadIdx.x;   // b*T*H
  int h  = idx & (NHEAD - 1);
  int bt = idx >> 4;
  int t  = bt % TT, b = bt / TT;
  const bf16* src = in + (size_t)bt * DDIM + h * HDIM;
  float v[HDIM];
#pragma unroll
  for (int i = 0; i < HDIM; ++i) v[i] = (float)src[i];
#pragma unroll
  for (int i = 0; i < 16; ++i) {
    float fr  = powf(10000.0f, -(float)i / 16.0f);
    float ang = (float)t * fr;
    float c = cosf(ang), s = sinf(ang);
    float x1 = v[i], x2 = v[i + 16];
    v[i]      = x1 * c - x2 * s;
    v[i + 16] = x2 * c + x1 * s;
  }
  size_t bh = (size_t)(b * NHEAD + h);
  if (vt == 0) {
    bf16* dst = out + (bh * TT + t) * HDIM;
#pragma unroll
    for (int i = 0; i < HDIM; ++i) dst[i] = (bf16)v[i];
  } else {
#pragma unroll
    for (int i = 0; i < HDIM; ++i)
      out[(bh * HDIM + i) * TT + t] = (bf16)v[i];
  }
}

// ---------------------------------------------------------------------------
// BF16 WMMA GEMM: C[M,N] = A[M,K] * Wt[N,K]^T (+bias)(gelu)(+resid)
// Block tile 64x64, 8 waves, each wave 16x32; K-step 32.
// Double-buffered async global->LDS staging (ASYNCcnt), b128 LDS frags.
// ---------------------------------------------------------------------------
__global__ __launch_bounds__(256)
void gemm_bf16_wmma(const bf16* __restrict__ A, const bf16* __restrict__ Wt,
                    float* __restrict__ Cf, bf16* __restrict__ Cb,
                    const float* __restrict__ bias, const float* __restrict__ resid,
                    int M, int N, int K, int act) {
  __shared__ __align__(16) bf16 As[2][64 * 32];   // [m][k]
  __shared__ __align__(16) bf16 Bs[2][64 * 40];   // [n][k], pitch 40
  const int tid = threadIdx.x, lane = tid & 31, wv = tid >> 5;
  const int wm = wv & 3, wn = wv >> 2;
  const int m0 = wm * 16;
  const int mblk = blockIdx.y * 64, nblk = blockIdx.x * 64;
  const int hl = lane >> 4;

  v8f acc0 = {}, acc1 = {};

  const int row8 = tid >> 2, col8 = (tid & 3) * 8;   // 64 rows x 32 cols

  const bf16* pA = A  + (size_t)(mblk + row8) * K + col8;
  const bf16* pB = Wt + (size_t)(nblk + row8) * K + col8;
  const unsigned lA0 = lds_addr(&As[0][row8 * 32 + col8]);
  const unsigned lA1 = lds_addr(&As[1][row8 * 32 + col8]);
  const unsigned lB0 = lds_addr(&Bs[0][row8 * 40 + col8]);
  const unsigned lB1 = lds_addr(&Bs[1][row8 * 40 + col8]);

  // prologue: tile 0 in flight
  async_copy16(lA0, pA);
  async_copy16(lB0, pB);

  int ib = 0;
  for (int kb = 0; kb < K; kb += 32, ib ^= 1) {
    wait_async0();
    __syncthreads();
    if (kb + 32 < K) {
      pA += 32; pB += 32;
      async_copy16(ib ? lA0 : lA1, pA);
      async_copy16(ib ? lB0 : lB1, pB);
      __builtin_prefetch(pA + 32, 0, 1);
      __builtin_prefetch(pB + 32, 0, 1);
    }
    const bf16* as = As[ib];
    const bf16* bs = Bs[ib];

    // A fragment (16x32): two contiguous 16B runs per lane
    int Mr = m0 + (lane & 15);
    v16bf af = cat8(*(const v8bf*)&as[Mr * 32 + hl * 16],
                    *(const v8bf*)&as[Mr * 32 + hl * 16 + 8]);
    // B fragments (32x16): contiguous K run per lane from [n][k] tile
    int Nc = wn * 32 + (lane & 15);
    int kk = hl * 16;
    v16bf bf0 = cat8(*(const v8bf*)&bs[Nc * 40 + kk],
                     *(const v8bf*)&bs[Nc * 40 + kk + 8]);
    v16bf bf1 = cat8(*(const v8bf*)&bs[(Nc + 16) * 40 + kk],
                     *(const v8bf*)&bs[(Nc + 16) * 40 + kk + 8]);
    acc0 = WMMA_BF16(af, bf0, acc0);
    acc1 = WMMA_BF16(af, bf1, acc1);
  }

  // epilogue
#pragma unroll
  for (int nt = 0; nt < 2; ++nt) {
    int gn = nblk + wn * 32 + nt * 16 + (lane & 15);
    float bv = bias ? bias[gn] : 0.0f;
#pragma unroll
    for (int r = 0; r < 8; ++r) {
      int gm = mblk + m0 + r + 8 * hl;
      float vv = (nt == 0 ? acc0[r] : acc1[r]) + bv;
      if (act == 1) vv = gelu_tanh(vv);
      if (resid) vv += resid[(size_t)gm * N + gn];
      if (Cf) Cf[(size_t)gm * N + gn] = vv;
      if (Cb) Cb[(size_t)gm * N + gn] = (bf16)vv;
    }
  }
}

// ---------------------------------------------------------------------------
// Causal flash attention with WMMA. 128 threads: 4 waves x 16 q-rows.
// q,k: (B,H,T,64) bf16.  v: (B,H,64,T) bf16 (channel-major).
// K/V tiles staged with double-buffered async global->LDS copies.
// out: (B,T,H*64) bf16.
// ---------------------------------------------------------------------------
__global__ __launch_bounds__(128)
void attn_kernel(const bf16* __restrict__ q, const bf16* __restrict__ k,
                 const bf16* __restrict__ v, bf16* __restrict__ out) {
  __shared__ __align__(16) bf16 Ks[2][32 * 72];     // [key][ch]
  __shared__ __align__(16) bf16 Vt[2][64 * 40];     // [ch][key]
  __shared__ __align__(16) bf16 Ps[4][16 * 40];     // per-wave P staging

  int bh = blockIdx.y;
  int b = bh >> 4, h = bh & 15;
  const bf16* qp = q + (size_t)bh * TT * HDIM;
  const bf16* kp = k + (size_t)bh * TT * HDIM;
  const bf16* vp = v + (size_t)bh * HDIM * TT;

  int tid = threadIdx.x, lane = tid & 31, wv = tid >> 5;
  int qrow0 = blockIdx.x * 64 + wv * 16;
  int hl = lane >> 4;

  // Q fragments: 4 x global_load_b128 per lane
  v16bf qa0, qa1;
  {
    int Mr = qrow0 + (lane & 15);
    const bf16* qr = qp + (size_t)Mr * HDIM;
    qa0 = cat8(*(const v8bf*)&qr[hl * 16],      *(const v8bf*)&qr[hl * 16 + 8]);
    qa1 = cat8(*(const v8bf*)&qr[32 + hl * 16], *(const v8bf*)&qr[32 + hl * 16 + 8]);
  }

  v8f acc[4] = {};
  float mrow[8], lrow[8];
#pragma unroll
  for (int r = 0; r < 8; ++r) { mrow[r] = -1e30f; lrow[r] = 0.0f; }

  const int jend = blockIdx.x * 64 + 63;
  const int ldr = tid >> 2, ldc = (tid & 3) * 16;   // K tile: 32 keys x 64 ch
  const int vc  = tid >> 1, vk  = (tid & 1) * 16;   // V tile: 64 ch x 32 keys

  const bf16* kg = kp + (size_t)ldr * HDIM + ldc;   // += 32*HDIM per step
  const bf16* vg = vp + (size_t)vc * TT + vk;       // += 32 per step
  const unsigned lK0 = lds_addr(&Ks[0][ldr * 72 + ldc]);
  const unsigned lK1 = lds_addr(&Ks[1][ldr * 72 + ldc]);
  const unsigned lV0 = lds_addr(&Vt[0][vc * 40 + vk]);
  const unsigned lV1 = lds_addr(&Vt[1][vc * 40 + vk]);

  // prologue: first K/V tile in flight (16B at +0 and +16 per lane)
  async_copy16(lK0, kg);  async_copy16_off16(lK0, kg);
  async_copy16(lV0, vg);  async_copy16_off16(lV0, vg);

  int ib = 0;
  for (int j0 = 0; j0 <= jend; j0 += 32, ib ^= 1) {
    wait_async0();
    __syncthreads();
    if (j0 + 32 <= jend) {
      kg += 32 * HDIM; vg += 32;
      unsigned nK = ib ? lK0 : lK1, nV = ib ? lV0 : lV1;
      async_copy16(nK, kg);  async_copy16_off16(nK, kg);
      async_copy16(nV, vg);  async_copy16_off16(nV, vg);
    }
    const bf16* ks = Ks[ib];
    const bf16* vt = Vt[ib];

    // S = Q K^T : two 16x16 key tiles (channel-contiguous B frags)
    v8f s0 = {}, s1 = {};
    {
      int nK = lane & 15;
      int cb = hl * 16;
      const bf16* kr0 = &ks[nK * 72];
      const bf16* kr1 = &ks[(nK + 16) * 72];
      v16bf b00 = cat8(*(const v8bf*)&kr0[cb],      *(const v8bf*)&kr0[cb + 8]);
      v16bf b01 = cat8(*(const v8bf*)&kr0[32 + cb], *(const v8bf*)&kr0[32 + cb + 8]);
      v16bf b10 = cat8(*(const v8bf*)&kr1[cb],      *(const v8bf*)&kr1[cb + 8]);
      v16bf b11 = cat8(*(const v8bf*)&kr1[32 + cb], *(const v8bf*)&kr1[32 + cb + 8]);
      s0 = WMMA_BF16(qa0, b00, s0);
      s0 = WMMA_BF16(qa1, b01, s0);
      s1 = WMMA_BF16(qa0, b10, s1);
      s1 = WMMA_BF16(qa1, b11, s1);
    }

    // causal mask, scale, online softmax update
    float sv0[8], sv1[8];
#pragma unroll
    for (int r = 0; r < 8; ++r) {
      int qi  = qrow0 + r + 8 * hl;
      int kj0 = j0 + (lane & 15);
      sv0[r] = (kj0      <= qi) ? s0[r] * 0.125f : -1e30f;
      sv1[r] = (kj0 + 16 <= qi) ? s1[r] * 0.125f : -1e30f;
    }
#pragma unroll
    for (int r = 0; r < 8; ++r) {
      float mx = fmaxf(sv0[r], sv1[r]);
#pragma unroll
      for (int off = 8; off > 0; off >>= 1) mx = fmaxf(mx, __shfl_xor(mx, off, 32));
      float mnew = fmaxf(mrow[r], mx);
      float p0 = expf(sv0[r] - mnew);
      float p1 = expf(sv1[r] - mnew);
      float rs = p0 + p1;
#pragma unroll
      for (int off = 8; off > 0; off >>= 1) rs += __shfl_xor(rs, off, 32);
      float scl = expf(mrow[r] - mnew);
      lrow[r] = lrow[r] * scl + rs;
      mrow[r] = mnew;
#pragma unroll
      for (int nt = 0; nt < 4; ++nt) acc[nt][r] *= scl;
      int prow = r + 8 * hl;
      Ps[wv][prow * 40 + (lane & 15)]      = (bf16)p0;
      Ps[wv][prow * 40 + 16 + (lane & 15)] = (bf16)p1;
    }

    // Re-fragment P (16x32) from wave-local LDS: 2 x ds_load_b128 per lane
    v16bf pa;
    {
      int Mr = lane & 15;
      pa = cat8(*(const v8bf*)&Ps[wv][Mr * 40 + hl * 16],
                *(const v8bf*)&Ps[wv][Mr * 40 + hl * 16 + 8]);
    }
    // O += P V : four 16x16 v-channel tiles (key-contiguous B frags)
    {
      int nc = lane & 15;
      int kk = hl * 16;
#pragma unroll
      for (int nt = 0; nt < 4; ++nt) {
        const bf16* vr = &vt[(nt * 16 + nc) * 40];
        v16bf bvf = cat8(*(const v8bf*)&vr[kk], *(const v8bf*)&vr[kk + 8]);
        acc[nt] = WMMA_BF16(pa, bvf, acc[nt]);
      }
    }
  }

  // normalize and write (B,T,H*64) bf16
#pragma unroll
  for (int r = 0; r < 8; ++r) {
    int qi = qrow0 + r + 8 * hl;
    float inv = 1.0f / lrow[r];
#pragma unroll
    for (int nt = 0; nt < 4; ++nt) {
      int col = nt * 16 + (lane & 15);
      out[((size_t)(b * TT + qi)) * DDIM + h * HDIM + col] = (bf16)(acc[nt][r] * inv);
    }
  }
}

// ---------------------------------------------------------------------------
extern "C" void kernel_launch(void* const* d_in, const int* in_sizes, int n_in,
                              void* d_out, int out_size, void* d_ws, size_t ws_size,
                              hipStream_t stream) {
  (void)in_sizes; (void)n_in; (void)out_size; (void)ws_size;
  const float* x       = (const float*)d_in[0];
  const float* w_q     = (const float*)d_in[1];
  const float* w_k     = (const float*)d_in[2];
  const float* w_v     = (const float*)d_in[3];
  const float* w_o     = (const float*)d_in[4];
  const float* alpha_p = (const float*)d_in[5];
  const float* delta_p = (const float*)d_in[6];
  const float* beta    = (const float*)d_in[7];
  const float* eta     = (const float*)d_in[8];
  const float* g1      = (const float*)d_in[9];
  const float* g2      = (const float*)d_in[10];
  const float* w_in    = (const float*)d_in[11];
  const float* b_in    = (const float*)d_in[12];
  const float* w_out   = (const float*)d_in[13];
  const float* b_out   = (const float*)d_in[14];

  const size_t BT = (size_t)BB * TT;       // 4096 rows
  char* ws = (char*)d_ws;
  size_t off = 0;
  auto alloc = [&](size_t bytes) -> char* {
    char* p = ws + off;
    off += (bytes + 255) & ~(size_t)255;
    return p;
  };

  float* xn_f   = (float*)alloc(BT * DDIM * 4);
  bf16*  xn_b   = (bf16*) alloc(BT * DDIM * 2);
  bf16*  y_b    = (bf16*) alloc(BT * DDIM * 2);
  bf16*  wq_t   = (bf16*) alloc((size_t)DDIM * DDIM * 2);
  bf16*  wk_t   = (bf16*) alloc((size_t)DDIM * DDIM * 2);
  bf16*  wv_t   = (bf16*) alloc((size_t)DDIM * DDIM * 2);
  bf16*  wo_t   = (bf16*) alloc((size_t)DDIM * DDIM * 2);
  bf16*  win_t  = (bf16*) alloc((size_t)DDIM * HIDC * 2);
  bf16*  wout_t = (bf16*) alloc((size_t)HIDC * DDIM * 2);
  bf16*  q_b    = (bf16*) alloc(BT * DDIM * 2);
  bf16*  k_b    = (bf16*) alloc(BT * DDIM * 2);
  bf16*  v_b    = (bf16*) alloc(BT * DDIM * 2);
  bf16*  qr     = (bf16*) alloc(BT * DDIM * 2);
  bf16*  kr     = (bf16*) alloc(BT * DDIM * 2);
  bf16*  vr_t   = (bf16*) alloc(BT * DDIM * 2);
  bf16*  attn_b = (bf16*) alloc(BT * DDIM * 2);
  float* x1_f   = (float*)alloc(BT * DDIM * 4);
  bf16*  xn2_b  = (bf16*) alloc(BT * DDIM * 2);
  bf16*  h_b    = (bf16*) alloc(BT * HIDC * 2);

  // weight transpose+convert: W (K,N) -> Wt (N,K) bf16
  {
    dim3 gsq(DDIM / 32, DDIM / 32);
    transpose_f32_to_bf16<<<gsq, 256, 0, stream>>>(w_q, wq_t, DDIM, DDIM);
    transpose_f32_to_bf16<<<gsq, 256, 0, stream>>>(w_k, wk_t, DDIM, DDIM);
    transpose_f32_to_bf16<<<gsq, 256, 0, stream>>>(w_v, wv_t, DDIM, DDIM);
    transpose_f32_to_bf16<<<gsq, 256, 0, stream>>>(w_o, wo_t, DDIM, DDIM);
    transpose_f32_to_bf16<<<dim3(HIDC / 32, DDIM / 32), 256, 0, stream>>>(w_in, win_t, DDIM, HIDC);
    transpose_f32_to_bf16<<<dim3(DDIM / 32, HIDC / 32), 256, 0, stream>>>(w_out, wout_t, HIDC, DDIM);
  }

  // xn = rmsnorm(x, g1)
  rmsnorm_kernel<<<(int)BT, 256, 0, stream>>>(x, g1, xn_f, xn_b);

  // y = ema(xn)
  ema_kernel<<<(BB * DDIM + 255) / 256, 256, 0, stream>>>(xn_f, alpha_p, delta_p, beta, eta, y_b);

  // q/k from y, v from xn
  dim3 gproj(DDIM / 64, (int)(BT / 64));
  gemm_bf16_wmma<<<gproj, 256, 0, stream>>>(y_b,  wq_t, nullptr, q_b, nullptr, nullptr, (int)BT, DDIM, DDIM, 0);
  gemm_bf16_wmma<<<gproj, 256, 0, stream>>>(y_b,  wk_t, nullptr, k_b, nullptr, nullptr, (int)BT, DDIM, DDIM, 0);
  gemm_bf16_wmma<<<gproj, 256, 0, stream>>>(xn_b, wv_t, nullptr, v_b, nullptr, nullptr, (int)BT, DDIM, DDIM, 0);

  // rope + head relayout (v channel-major)
  int nrope = BB * TT * NHEAD;
  rope_kernel<<<nrope / 256, 256, 0, stream>>>(q_b, qr, 0);
  rope_kernel<<<nrope / 256, 256, 0, stream>>>(k_b, kr, 0);
  rope_kernel<<<nrope / 256, 256, 0, stream>>>(v_b, vr_t, 1);

  // causal flash attention
  attn_kernel<<<dim3(TT / 64, BB * NHEAD), 128, 0, stream>>>(qr, kr, vr_t, attn_b);

  // x1 = x + attn @ w_o
  gemm_bf16_wmma<<<gproj, 256, 0, stream>>>(attn_b, wo_t, x1_f, nullptr, nullptr, x, (int)BT, DDIM, DDIM, 0);

  // xn2 = rmsnorm(x1, g2)
  rmsnorm_kernel<<<(int)BT, 256, 0, stream>>>(x1_f, g2, nullptr, xn2_b);

  // h = gelu(xn2 @ w_in + b_in)
  dim3 gffn1(HIDC / 64, (int)(BT / 64));
  gemm_bf16_wmma<<<gffn1, 256, 0, stream>>>(xn2_b, win_t, nullptr, h_b, b_in, nullptr, (int)BT, HIDC, DDIM, 1);

  // out = x1 + h @ w_out + b_out
  gemm_bf16_wmma<<<gproj, 256, 0, stream>>>(h_b, wout_t, (float*)d_out, nullptr, b_out, x1_f, (int)BT, DDIM, HIDC, 0);
}
